// GLOM_41455024341677
// MI455X (gfx1250) — compile-verified, matrix operand order checked by reading, not run on hardware
//
#include <hip/hip_runtime.h>
#include <cstddef>

typedef __attribute__((ext_vector_type(16))) _Float16 v16h;
typedef __attribute__((ext_vector_type(8)))  _Float16 v8h;
typedef __attribute__((ext_vector_type(4)))  _Float16 v4h;
typedef __attribute__((ext_vector_type(8)))  float    v8f;

#define TPB 256
#define LDA 40   // padded LDS row stride in halves (80B: 16B-aligned, conflict-free)

// ---------------------------------------------------------------------------
// WMMA fragment load from an LDS tile laid out [row][k], stride LDA halves.
// CDNA5 16-bit A/B layout: lanes 0-15 hold row=lane, K={0..7,16..23};
// lanes 16-31 hold row=lane-16, K={8..15,24..31}. Two 16B ds reads.
// ---------------------------------------------------------------------------
__device__ __forceinline__ v16h load_frag(const _Float16* base, int row0, int lane) {
  int l15 = lane & 15, hi = (lane >> 4) & 1;
  const _Float16* r = base + (row0 + l15) * LDA;
  v8h lo = *(const v8h*)(r + hi * 8);
  v8h up = *(const v8h*)(r + 16 + hi * 8);
  return __builtin_shufflevector(lo, up, 0,1,2,3,4,5,6,7,8,9,10,11,12,13,14,15);
}

__device__ __forceinline__ v8f wmma_f16(v16h a, v16h b, v8f c) {
  return __builtin_amdgcn_wmma_f32_16x16x32_f16(false, a, false, b, (short)0, c, false, false);
}

__device__ __forceinline__ void cvt_store4(_Float16* dst, float4 v) {
  v4h h = { (_Float16)v.x, (_Float16)v.y, (_Float16)v.z, (_Float16)v.w };
  *(v4h*)dst = h;
}

// ---------------------------------------------------------------------------
// Utility kernels
// ---------------------------------------------------------------------------
__global__ void k_zero(float* p, int n) {
  int i = blockIdx.x * blockDim.x + threadIdx.x;
  if (i < n) p[i] = 0.f;
}

// NeRF positional table: loc[hw][40] = [pe_h(20) | pe_w(20)], band-interleaved sin/cos
__global__ void k_loc(float* loc) {
  int idx = blockIdx.x * blockDim.x + threadIdx.x;
  if (idx >= 1024 * 40) return;
  int hw = idx / 40, j = idx % 40;
  int h = hw >> 5, w = hw & 31;
  int jj = (j < 20) ? j : j - 20;
  float p = (j < 20) ? (2.f * h / 32.f - 1.f) : (2.f * w / 32.f - 1.f);
  float ang = exp2f((float)(jj >> 1)) * 3.14159265358979323846f * p;
  loc[idx] = (jj & 1) ? cosf(ang) : sinf(ang);
}

// ---------------------------------------------------------------------------
// Patch conv as implicit-im2col GEMM: z[m,e] = sum_k x(m,k)*Wc[e,k] + bc[e]
// m=(b,h,w) in 8192, k=(c,kh,kw) in 192, e in 512.  BM=128 BN=64, 2x2 tiles/wave.
// ---------------------------------------------------------------------------
__global__ void k_conv(const float* __restrict__ x, const float* __restrict__ wc,
                       const float* __restrict__ bc, float* __restrict__ z) {
  __shared__ __align__(16) _Float16 As[128 * LDA];
  __shared__ __align__(16) _Float16 Bs[64 * LDA];
  int tid = threadIdx.x, lane = tid & 31, wv = tid >> 5;
  int m0 = blockIdx.y * 128, n0 = blockIdx.x * 64;
  int swm = (wv >> 1) * 32, swn = (wv & 1) * 32;
  v8f c00 = {}, c01 = {}, c10 = {}, c11 = {};
  for (int k0 = 0; k0 < 192; k0 += 32) {
    float4 ta[4], tb[2];
#pragma unroll
    for (int r = 0; r < 4; ++r) {
      int u = tid + r * 256, im = u >> 3, kq = (u & 7) * 4;
      int m = m0 + im, k = k0 + kq;
      int b = m >> 10, hw = m & 1023, h = hw >> 5, w = hw & 31;
      int ch = k >> 6, rr = k & 63, kh = rr >> 3, kw = rr & 7;
      ta[r] = *(const float4*)&x[((b * 3 + ch) * 256 + (h * 8 + kh)) * 256 + (w * 8 + kw)];
    }
#pragma unroll
    for (int r = 0; r < 2; ++r) {
      int u = tid + r * 256, jn = u >> 3, kq = (u & 7) * 4;
      tb[r] = *(const float4*)&wc[(n0 + jn) * 192 + (k0 + kq)];
    }
#pragma unroll
    for (int r = 0; r < 4; ++r) {
      int u = tid + r * 256, im = u >> 3, kq = (u & 7) * 4;
      cvt_store4(&As[im * LDA + kq], ta[r]);
    }
#pragma unroll
    for (int r = 0; r < 2; ++r) {
      int u = tid + r * 256, jn = u >> 3, kq = (u & 7) * 4;
      cvt_store4(&Bs[jn * LDA + kq], tb[r]);
    }
    __syncthreads();
    v16h a0 = load_frag(As, swm, lane), a1 = load_frag(As, swm + 16, lane);
    v16h b0 = load_frag(Bs, swn, lane), b1 = load_frag(Bs, swn + 16, lane);
    c00 = wmma_f16(a0, b0, c00);  c01 = wmma_f16(a0, b1, c01);
    c10 = wmma_f16(a1, b0, c10);  c11 = wmma_f16(a1, b1, c11);
    __syncthreads();
  }
  int l15 = lane & 15, hi8 = ((lane >> 4) & 1) * 8;
  int col0 = n0 + swn + l15, col1 = col0 + 16;
  int row0 = m0 + swm + hi8,  row1 = row0 + 16;
  float bi0 = bc[col0], bi1 = bc[col1];
#pragma unroll
  for (int r = 0; r < 8; ++r) {
    z[(row0 + r) * 512 + col0] = c00[r] + bi0;
    z[(row0 + r) * 512 + col1] = c01[r] + bi1;
    z[(row1 + r) * 512 + col0] = c10[r] + bi0;
    z[(row1 + r) * 512 + col1] = c11[r] + bi1;
  }
}

// ---------------------------------------------------------------------------
// BN stats: per-channel sum & sumsq over 8192 rows (atomic accumulate)
// ---------------------------------------------------------------------------
__global__ void k_bn_stats(const float* __restrict__ xin, float* __restrict__ stats) {
  int row0 = blockIdx.x * 32, tid = threadIdx.x;
  for (int c = tid; c < 512; c += TPB) {
    float s = 0.f, s2 = 0.f;
#pragma unroll 8
    for (int r = 0; r < 32; ++r) {
      float v = xin[(row0 + r) * 512 + c];
      s += v; s2 += v * v;
    }
    atomicAdd(&stats[c], s);
    atomicAdd(&stats[512 + c], s2);
  }
}

// In-place BN, seed new[l]=w1*y, accumulate post-BN total sum (attention gate)
__global__ void k_bn_apply(float* __restrict__ xio, float* __restrict__ nw,
                           const float* __restrict__ stats, const float* __restrict__ g,
                           const float* __restrict__ bta, const float* __restrict__ w1p,
                           float* __restrict__ gate) {
  const float invM = 1.f / 8192.f;
  float w1 = *w1p;
  int base = (blockIdx.x * blockDim.x + threadIdx.x) * 4;
  int c0 = base & 511;
  float4 xv = *(const float4*)&xio[base];
  float y[4], s = 0.f;
  float xa[4] = { xv.x, xv.y, xv.z, xv.w };
#pragma unroll
  for (int i = 0; i < 4; ++i) {
    int c = c0 + i;
    float m = stats[c] * invM;
    float v = stats[512 + c] * invM - m * m;
    y[i] = (xa[i] - m) * rsqrtf(v + 1e-5f) * g[c] + bta[c];
    s += y[i];
  }
  float4 yv = { y[0], y[1], y[2], y[3] };
  float4 nv = { w1 * y[0], w1 * y[1], w1 * y[2], w1 * y[3] };
  *(float4*)&xio[base] = yv;
  *(float4*)&nw[base]  = nv;
  for (int o = 16; o > 0; o >>= 1) s += __shfl_xor(s, o, 32);
  if ((threadIdx.x & 31) == 0) atomicAdd(gate, s);
}

// ---------------------------------------------------------------------------
// BU/TD linear: dst[m,n] += coef*( [src|loc](m,k) @ W[n,k] + bias[n] )
// K=552 zero-padded to 576 (18 K-steps).  BM=128 BN=64.
// ---------------------------------------------------------------------------
__global__ void k_lin(const float* __restrict__ src, const float* __restrict__ loc,
                      const float* __restrict__ W, const float* __restrict__ bias,
                      const float* __restrict__ coefp, float* __restrict__ dst) {
  __shared__ __align__(16) _Float16 As[128 * LDA];
  __shared__ __align__(16) _Float16 Bs[64 * LDA];
  int tid = threadIdx.x, lane = tid & 31, wv = tid >> 5;
  int m0 = blockIdx.y * 128, n0 = blockIdx.x * 64;
  int swm = (wv >> 1) * 32, swn = (wv & 1) * 32;
  v8f c00 = {}, c01 = {}, c10 = {}, c11 = {};
  const float4 zf4 = { 0.f, 0.f, 0.f, 0.f };
  for (int k0 = 0; k0 < 576; k0 += 32) {
    float4 ta[4], tb[2];
#pragma unroll
    for (int r = 0; r < 4; ++r) {
      int u = tid + r * 256, im = u >> 3, kq = (u & 7) * 4;
      int m = m0 + im, k = k0 + kq;
      if (k < 512)      ta[r] = *(const float4*)&src[m * 512 + k];
      else if (k < 552) ta[r] = *(const float4*)&loc[(m & 1023) * 40 + (k - 512)];
      else              ta[r] = zf4;
    }
#pragma unroll
    for (int r = 0; r < 2; ++r) {
      int u = tid + r * 256, jn = u >> 3, kq = (u & 7) * 4;
      int k = k0 + kq;
      tb[r] = (k < 552) ? *(const float4*)&W[(n0 + jn) * 552 + k] : zf4;
    }
#pragma unroll
    for (int r = 0; r < 4; ++r) {
      int u = tid + r * 256, im = u >> 3, kq = (u & 7) * 4;
      cvt_store4(&As[im * LDA + kq], ta[r]);
    }
#pragma unroll
    for (int r = 0; r < 2; ++r) {
      int u = tid + r * 256, jn = u >> 3, kq = (u & 7) * 4;
      cvt_store4(&Bs[jn * LDA + kq], tb[r]);
    }
    __syncthreads();
    v16h a0 = load_frag(As, swm, lane), a1 = load_frag(As, swm + 16, lane);
    v16h b0 = load_frag(Bs, swn, lane), b1 = load_frag(Bs, swn + 16, lane);
    c00 = wmma_f16(a0, b0, c00);  c01 = wmma_f16(a0, b1, c01);
    c10 = wmma_f16(a1, b0, c10);  c11 = wmma_f16(a1, b1, c11);
    __syncthreads();
  }
  float coef = *coefp;
  int l15 = lane & 15, hi8 = ((lane >> 4) & 1) * 8;
  int col0 = n0 + swn + l15, col1 = col0 + 16;
  int row0 = m0 + swm + hi8,  row1 = row0 + 16;
  float bi0 = bias[col0], bi1 = bias[col1];
#pragma unroll
  for (int r = 0; r < 8; ++r) {
    dst[(row0 + r) * 512 + col0] += coef * (c00[r] + bi0);
    dst[(row0 + r) * 512 + col1] += coef * (c01[r] + bi1);
    dst[(row1 + r) * 512 + col0] += coef * (c10[r] + bi0);
    dst[(row1 + r) * 512 + col1] += coef * (c11[r] + bi1);
  }
}

// ---------------------------------------------------------------------------
// Attention pass 1: ex[b,q,k] = exp(BETA * e_b[q,:]·e_b[k,:]) stored f16,
// plus denom[q] += row sums (accumulated across batch AND key tiles).
// ---------------------------------------------------------------------------
__global__ void k_scores(const float* __restrict__ e, _Float16* __restrict__ ex,
                         float* __restrict__ denom) {
  __shared__ __align__(16) _Float16 As[128 * LDA];
  __shared__ __align__(16) _Float16 Bs[64 * LDA];
  int tid = threadIdx.x, lane = tid & 31, wv = tid >> 5;
  int m0 = blockIdx.y * 128, n0 = blockIdx.x * 64, b = blockIdx.z;
  const float* eb = e + (size_t)b * 1024 * 512;
  int swm = (wv >> 1) * 32, swn = (wv & 1) * 32;
  v8f c00 = {}, c01 = {}, c10 = {}, c11 = {};
  for (int k0 = 0; k0 < 512; k0 += 32) {
    float4 ta[4], tb[2];
#pragma unroll
    for (int r = 0; r < 4; ++r) {
      int u = tid + r * 256, im = u >> 3, kq = (u & 7) * 4;
      ta[r] = *(const float4*)&eb[(m0 + im) * 512 + (k0 + kq)];
    }
#pragma unroll
    for (int r = 0; r < 2; ++r) {
      int u = tid + r * 256, jn = u >> 3, kq = (u & 7) * 4;
      tb[r] = *(const float4*)&eb[(n0 + jn) * 512 + (k0 + kq)];
    }
#pragma unroll
    for (int r = 0; r < 4; ++r) {
      int u = tid + r * 256, im = u >> 3, kq = (u & 7) * 4;
      cvt_store4(&As[im * LDA + kq], ta[r]);
    }
#pragma unroll
    for (int r = 0; r < 2; ++r) {
      int u = tid + r * 256, jn = u >> 3, kq = (u & 7) * 4;
      cvt_store4(&Bs[jn * LDA + kq], tb[r]);
    }
    __syncthreads();
    v16h a0 = load_frag(As, swm, lane), a1 = load_frag(As, swm + 16, lane);
    v16h b0 = load_frag(Bs, swn, lane), b1 = load_frag(Bs, swn + 16, lane);
    c00 = wmma_f16(a0, b0, c00);  c01 = wmma_f16(a0, b1, c01);
    c10 = wmma_f16(a1, b0, c10);  c11 = wmma_f16(a1, b1, c11);
    __syncthreads();
  }
  int l15 = lane & 15, hi8 = ((lane >> 4) & 1) * 8;
  int col0 = n0 + swn + l15, col1 = col0 + 16;
  int row0 = m0 + swm + hi8,  row1 = row0 + 16;
  size_t bb = (size_t)b * 1024;
  float rs0[8], rs1[8];
#pragma unroll
  for (int r = 0; r < 8; ++r) {
    float v00 = expf(0.001f * c00[r]);
    float v01 = expf(0.001f * c01[r]);
    float v10 = expf(0.001f * c10[r]);
    float v11 = expf(0.001f * c11[r]);
    ex[(bb + row0 + r) * 1024 + col0] = (_Float16)v00;
    ex[(bb + row0 + r) * 1024 + col1] = (_Float16)v01;
    ex[(bb + row1 + r) * 1024 + col0] = (_Float16)v10;
    ex[(bb + row1 + r) * 1024 + col1] = (_Float16)v11;
    rs0[r] = v00 + v01;
    rs1[r] = v10 + v11;
  }
  // reduce across the 16 lanes sharing the same rows (xor masks < 16)
  for (int o = 1; o < 16; o <<= 1)
#pragma unroll
    for (int r = 0; r < 8; ++r) {
      rs0[r] += __shfl_xor(rs0[r], o, 32);
      rs1[r] += __shfl_xor(rs1[r], o, 32);
    }
  if (l15 == 0)
#pragma unroll
    for (int r = 0; r < 8; ++r) {
      atomicAdd(&denom[row0 + r], rs0[r]);
      atomicAdd(&denom[row1 + r], rs1[r]);
    }
}

// ---------------------------------------------------------------------------
// Attention pass 2: dst[b,q,:] += w4*gate/denom[q] * (ex[b,q,:] @ e_b)
// A is f16 already; B is K×N row-major -> transpose into LDS during staging.
// ---------------------------------------------------------------------------
__global__ void k_avg(const _Float16* __restrict__ ex, const float* __restrict__ e,
                      const float* __restrict__ denom, const float* __restrict__ gsum,
                      const float* __restrict__ w4p, float* __restrict__ dst) {
  __shared__ __align__(16) _Float16 As[128 * LDA];
  __shared__ __align__(16) _Float16 Bs[64 * LDA];
  int tid = threadIdx.x, lane = tid & 31, wv = tid >> 5;
  int m0 = blockIdx.y * 128, n0 = blockIdx.x * 64, b = blockIdx.z;
  const _Float16* exb = ex + (size_t)b * 1024 * 1024;
  const float*    eb  = e  + (size_t)b * 1024 * 512;
  int swm = (wv >> 1) * 32, swn = (wv & 1) * 32;
  v8f c00 = {}, c01 = {}, c10 = {}, c11 = {};
  for (int k0 = 0; k0 < 1024; k0 += 32) {
    v8h tha[2];
    float4 tb[2];
#pragma unroll
    for (int r = 0; r < 2; ++r) {
      int u = tid + r * 256, im = u >> 2, ko = (u & 3) * 8;
      tha[r] = *(const v8h*)&exb[(size_t)(m0 + im) * 1024 + (k0 + ko)];
    }
#pragma unroll
    for (int r = 0; r < 2; ++r) {
      int u = tid + r * 256, kk = u >> 4, nq = (u & 15) * 4;
      tb[r] = *(const float4*)&eb[(k0 + kk) * 512 + (n0 + nq)];
    }
#pragma unroll
    for (int r = 0; r < 2; ++r) {
      int u = tid + r * 256, im = u >> 2, ko = (u & 3) * 8;
      *(v8h*)&As[im * LDA + ko] = tha[r];
    }
#pragma unroll
    for (int r = 0; r < 2; ++r) {      // transposed scatter into Bs[n][k]
      int u = tid + r * 256, kk = u >> 4, nq = (u & 15) * 4;
      Bs[(nq + 0) * LDA + kk] = (_Float16)tb[r].x;
      Bs[(nq + 1) * LDA + kk] = (_Float16)tb[r].y;
      Bs[(nq + 2) * LDA + kk] = (_Float16)tb[r].z;
      Bs[(nq + 3) * LDA + kk] = (_Float16)tb[r].w;
    }
    __syncthreads();
    v16h a0 = load_frag(As, swm, lane), a1 = load_frag(As, swm + 16, lane);
    v16h b0 = load_frag(Bs, swn, lane), b1 = load_frag(Bs, swn + 16, lane);
    c00 = wmma_f16(a0, b0, c00);  c01 = wmma_f16(a0, b1, c01);
    c10 = wmma_f16(a1, b0, c10);  c11 = wmma_f16(a1, b1, c11);
    __syncthreads();
  }
  float gate = (*gsum != 0.f) ? 1.f : 0.f;
  float w4 = *w4p * gate;
  int l15 = lane & 15, hi8 = ((lane >> 4) & 1) * 8;
  int col0 = n0 + swn + l15, col1 = col0 + 16;
  int row0 = m0 + swm + hi8,  row1 = row0 + 16;
  int bb = b * 1024;
#pragma unroll
  for (int r = 0; r < 8; ++r) {
    float s0 = w4 / denom[row0 + r];
    float s1 = w4 / denom[row1 + r];
    dst[(bb + row0 + r) * 512 + col0] += s0 * c00[r];
    dst[(bb + row0 + r) * 512 + col1] += s0 * c01[r];
    dst[(bb + row1 + r) * 512 + col0] += s1 * c10[r];
    dst[(bb + row1 + r) * 512 + col1] += s1 * c11[r];
  }
}

// ---------------------------------------------------------------------------
// Final layout: emb[l][b,hw,e] -> out[b,e,hw,l]
// ---------------------------------------------------------------------------
__global__ void k_out(const float* __restrict__ emb, float* __restrict__ out) {
  size_t idx = (size_t)blockIdx.x * blockDim.x + threadIdx.x;
  const size_t total = 5ull * 8 * 1024 * 512;
  if (idx >= total) return;
  int l = (int)(idx % 5); size_t r = idx / 5;
  int hw = (int)(r % 1024); r /= 1024;
  int ec = (int)(r % 512);
  int b  = (int)(r / 512);
  out[idx] = emb[(((size_t)l * 8 + b) * 1024 + hw) * 512 + ec];
}

// ---------------------------------------------------------------------------
extern "C" void kernel_launch(void* const* d_in, const int* in_sizes, int n_in,
                              void* d_out, int out_size, void* d_ws, size_t ws_size,
                              hipStream_t stream) {
  (void)in_sizes; (void)n_in; (void)out_size; (void)ws_size;
  const float* x      = (const float*)d_in[0];
  const float* conv_w = (const float*)d_in[1];
  const float* conv_b = (const float*)d_in[2];
  const float* bu_w   = (const float*)d_in[3];
  const float* bu_b   = (const float*)d_in[4];
  const float* td_w   = (const float*)d_in[5];
  const float* td_b   = (const float*)d_in[6];
  const float* bn_g   = (const float*)d_in[7];
  const float* bn_b   = (const float*)d_in[8];
  const float* w1     = (const float*)d_in[9];
  const float* w2     = (const float*)d_in[10];
  const float* w3     = (const float*)d_in[11];
  const float* w4     = (const float*)d_in[12];

  const size_t LVL = 8ull * 1024 * 512;          // floats per level tensor
  char* ws = (char*)d_ws;
  float*    bufA  = (float*)ws;                                  // 5*LVL f32
  float*    bufB  = bufA + 5 * LVL;                              // 5*LVL f32
  _Float16* exb   = (_Float16*)(ws + 10 * LVL * sizeof(float));  // 8M halves
  float*    loc   = (float*)((char*)exb + 8ull * 1024 * 1024 * 2);
  float*    denom = loc + 1024 * 40;                             // 1024
  float*    stats = denom + 1024;                                // 1025 (sum|sumsq|gate)

  dim3 blk(TPB);

  k_loc<<<(1024 * 40 + TPB - 1) / TPB, blk, 0, stream>>>(loc);
  // prev levels 1..4 start at zero
  k_zero<<<(int)((4 * LVL + TPB - 1) / TPB), blk, 0, stream>>>(bufA + LVL, (int)(4 * LVL));
  // conv -> prev level 0
  k_conv<<<dim3(8, 64), blk, 0, stream>>>(x, conv_w, conv_b, bufA);

  float* prev = bufA;
  float* nw   = bufB;
  for (int t = 0; t < 2; ++t) {
    for (int l = 0; l < 5; ++l) {
      float* pl = prev + (size_t)l * LVL;
      float* nl = nw   + (size_t)l * LVL;
      k_zero<<<(1025 + TPB - 1) / TPB, blk, 0, stream>>>(stats, 1025);
      k_bn_stats<<<256, blk, 0, stream>>>(pl, stats);
      k_bn_apply<<<4096, blk, 0, stream>>>(pl, nl, stats, bn_g + l * 512, bn_b + l * 512,
                                           w1, stats + 1024);
      if (l > 0)  // bottom-up reads prev[l-1] already BN'd this cycle
        k_lin<<<dim3(8, 64), blk, 0, stream>>>(prev + (size_t)(l - 1) * LVL, loc,
            bu_w + (size_t)l * 512 * 552, bu_b + l * 512, w2, nl);
      if (l < 4)  // top-down reads prev[l+1] NOT yet BN'd this cycle
        k_lin<<<dim3(8, 64), blk, 0, stream>>>(prev + (size_t)(l + 1) * LVL, loc,
            td_w + (size_t)l * 512 * 552, td_b + l * 512, w3, nl);
      k_zero<<<(1024 + TPB - 1) / TPB, blk, 0, stream>>>(denom, 1024);
      k_scores<<<dim3(16, 8, 8), blk, 0, stream>>>(pl, exb, denom);
      k_avg<<<dim3(8, 8, 8), blk, 0, stream>>>(exb, pl, denom, stats + 1024, w4, nl);
    }
    float* tmp = prev; prev = nw; nw = tmp;
  }
  // after the final swap, `prev` holds the last cycle's embeddings
  k_out<<<(int)((5 * LVL + TPB - 1) / TPB), blk, 0, stream>>>(prev, (float*)d_out);
}
